// SeqLabeling_29910152249580
// MI455X (gfx1250) — matrix operand order repaired and checked by reading, not compiled, not run on hardware
//
#include <hip/hip_runtime.h>
#include <cstddef>

#define B_BATCH 256
#define T_LEN   1024
#define L_LAB   64
#define NS      66          // real states (L+2)
#define PW      104         // p_s (f16) row stride, 16B-aligned, conflict-free
#define PREVW   100         // prev_s (f32) row stride
#define SMALLV  (-1000.0f)
#define L2E     1.4426950408889634f   // log2(e)
#define LN2     0.6931471805599453f   // ln(2)

typedef _Float16 v8h  __attribute__((ext_vector_type(8)));
typedef _Float16 v16h __attribute__((ext_vector_type(16)));
typedef float    v8f  __attribute__((ext_vector_type(8)));

// Raw hardware transcendentals (base-2). Safe here: log inputs are >= ~0.5
// (normal), exp2 inputs are finite and <= 0, so no denorm/range fixups needed.
__device__ __forceinline__ float fast_exp(float x) {   // e^x
  return __builtin_amdgcn_exp2f(x * L2E);
}
__device__ __forceinline__ float fast_log(float x) {   // ln(x), x normal
  return __builtin_amdgcn_logf(x) * LN2;
}

// ---------------------------------------------------------------------------
// Forward algorithm: per step, D = P x E with P = exp(prev - rowmax) (f16),
// E = exp(transitions) (f16, constant, held in registers as WMMA B fragments).
// new = ob + rowmax + log(D). One WG = 16 batch rows, 5 waves = 5 N-tiles.
// ---------------------------------------------------------------------------
__global__ __launch_bounds__(160)
void crf_forward_kernel(const float* __restrict__ pred,
                        const float* __restrict__ trans,
                        const int*   __restrict__ seqlen,
                        float*       __restrict__ score_ws)
{
  __shared__ __align__(16) float    prev_s[16 * PREVW];
  __shared__ __align__(16) _Float16 p_s[16 * PW];
  __shared__ float  rmax_s[16];
  __shared__ int    slen_s[16];

  const int tid   = threadIdx.x;
  const int wave  = tid >> 5;            // 0..4  -> N-tile
  const int lane  = tid & 31;
  const int m0    = blockIdx.x << 4;     // first batch row of this WG
  const int nloc  = lane & 15;
  const int n     = (wave << 4) + nloc;  // output state column (0..79)
  const int mhalf = (lane < 16) ? 0 : 8; // C/D layout: rows 0-7 vs 8-15
  const int m_a   = lane & 15;           // A layout: M row = lane%16
  const int kb0   = (lane < 16) ? 0 : 8; // A layout K base

  if (tid < 16) slen_s[tid] = seqlen[m0 + tid];
  // prev0 = b_s : state 64 (start) = 0, everything else (incl. K-pad) = SMALL
  for (int i = tid; i < 16 * PREVW; i += 160)
    prev_s[i] = ((i % PREVW) == L_LAB) ? 0.0f : SMALLV;

  // Constant B operand: E = exp(transitions), three 32x16 K-chunks.
  // B layout (16-bit, wave32): element i holds K=c*32+kb0+i (i<8) or c*32+16+kb0+(i-8) (i>=8),
  // column N striped over lane%16.
  v16h bfrag[3];
#pragma unroll
  for (int c = 0; c < 3; ++c) {
    v16h bf;
#pragma unroll
    for (int i = 0; i < 16; ++i) {
      const int k = c * 32 + kb0 + i + ((i < 8) ? 0 : 8);
      const float e = (k < NS && n < NS) ? fast_exp(trans[k * NS + n]) : 0.0f;
      bf[i] = (_Float16)e;
    }
    bfrag[c] = bf;
  }
  __syncthreads();

  for (int step = 0; step <= T_LEN + 1; ++step) {
    const bool do_mm = (step <= T_LEN);   // 1025 scan steps; last pass = capture only

    // ---- hoisted observation loads (overlap global latency with phase 1) ----
    float ob_r[8];
    if (do_mm && n < NS) {
#pragma unroll
      for (int r = 0; r < 8; ++r) {
        const int m = mhalf + r;
        const int b = m0 + m;
        float ob;
        if (step < T_LEN) {
          ob = (n < L_LAB) ? pred[((size_t)b * T_LEN + step) * L_LAB + n] : SMALLV;
          if (n < L_LAB && step + 1 < T_LEN)
            __builtin_prefetch(&pred[((size_t)b * T_LEN + step + 1) * L_LAB + n], 0, 0);
        } else {
          ob = (n < L_LAB) ? 0.0f : SMALLV;   // r_p row
        }
        if (step == slen_s[m])                // e_s row (obs index seq_len+1)
          ob += (n < L_LAB) ? SMALLV : ((n == L_LAB) ? 0.0f : 1000.0f);
        ob_r[r] = ob;
      }
    }

    // ---- Phase 1: rowmax, p = exp(prev - max) -> f16 LDS, rowsum ----
    for (int m = wave; m < 16; m += 5) {
      const float* row = prev_s + m * PREVW;
      const float v0 = row[lane];
      const float v1 = row[lane + 32];
      const float v2 = row[lane + 64];
      float mx = fmaxf(v0, fmaxf(v1, v2));
#pragma unroll
      for (int off = 16; off > 0; off >>= 1)
        mx = fmaxf(mx, __shfl_xor(mx, off, 32));
      const float p0 = fast_exp(v0 - mx);
      const float p1 = fast_exp(v1 - mx);
      const float p2 = fast_exp(v2 - mx);
      _Float16* prow = p_s + m * PW;
      prow[lane]      = (_Float16)p0;
      prow[lane + 32] = (_Float16)p1;
      prow[lane + 64] = (_Float16)p2;
      float sm = p0 + p1 + p2;               // logsumexp(prev) = mx + log(sm)
#pragma unroll
      for (int off = 16; off > 0; off >>= 1)
        sm += __shfl_xor(sm, off, 32);
      if (lane == 0) {
        rmax_s[m] = mx;
        if (step == slen_s[m] + 1)           // prev == alpha after step seq_len
          score_ws[m0 + m] = mx + fast_log(sm);
      }
    }
    __syncthreads();
    if (!do_mm) break;

    // ---- Phase 2: D = P x E via v_wmma_f32_16x16x32_f16 ----
    // Preload all three A fragments so the six ds_load_b128 overlap, then run
    // the three WMMAs back-to-back (D->C accumulate chain: no hazard NOPs).
    v16h afrag[3];
#pragma unroll
    for (int c = 0; c < 3; ++c) {
      const _Float16* rp = p_s + m_a * PW + c * 32 + kb0;
      const v8h lo = *(const v8h*)(rp);        // K = base+0..7   (ds_load_b128)
      const v8h hi = *(const v8h*)(rp + 16);   // K = base+16..23 (ds_load_b128)
      afrag[c] = __builtin_shufflevector(lo, hi,
          0,1,2,3,4,5,6,7,8,9,10,11,12,13,14,15);
    }
    v8f acc = {};
#pragma unroll
    for (int c = 0; c < 3; ++c)
      acc = __builtin_amdgcn_wmma_f32_16x16x32_f16(false, afrag[c], false, bfrag[c],
                                                   (short)0, acc, false, false);

    if (n < NS) {
#pragma unroll
      for (int r = 0; r < 8; ++r) {
        const int m = mhalf + r;
        prev_s[m * PREVW + n] = ob_r[r] + rmax_s[m] + fast_log(acc[r]);
      }
    }
    __syncthreads();
  }
}

// ---------------------------------------------------------------------------
// Gold path score per batch: emissions + transitions along the reference path.
// ---------------------------------------------------------------------------
__global__ __launch_bounds__(256)
void crf_gold_kernel(const float* __restrict__ pred,
                     const float* __restrict__ trans,
                     const int*   __restrict__ ref,
                     const int*   __restrict__ seqlen,
                     float*       __restrict__ gold_ws)
{
  __shared__ float red[256];
  const int b  = blockIdx.x;
  const int sl = seqlen[b];
  const int t0 = threadIdx.x;
  const float* predb = pred + (size_t)b * T_LEN * L_LAB;
  const int*   refb  = ref + b * T_LEN;

  float acc = 0.0f;
  for (int t = t0; t < sl; t += 256)                 // gold emissions
    acc += predb[(size_t)t * L_LAB + refb[t]];
  for (int t = t0; t <= sl; t += 256) {              // gold transitions
    const int from = (t == 0)  ? L_LAB       : refb[t - 1];   // 64 = start
    const int to   = (t == sl) ? (L_LAB + 1) : refb[t];       // 65 = end
    acc += trans[from * NS + to];
  }
  red[t0] = acc;
  __syncthreads();
  for (int s = 128; s > 0; s >>= 1) {
    if (t0 < s) red[t0] += red[t0 + s];
    __syncthreads();
  }
  if (t0 == 0) gold_ws[b] = red[0];
}

// ---------------------------------------------------------------------------
// Deterministic final reduction: sum(score) - sum(gold) -> scalar output.
// ---------------------------------------------------------------------------
__global__ __launch_bounds__(256)
void crf_reduce_kernel(const float* __restrict__ ws, float* __restrict__ out)
{
  __shared__ float red[256];
  const int t = threadIdx.x;
  red[t] = ws[t] - ws[256 + t];
  __syncthreads();
  for (int s = 128; s > 0; s >>= 1) {
    if (t < s) red[t] += red[t + s];
    __syncthreads();
  }
  if (t == 0) out[0] = red[0];
}

extern "C" void kernel_launch(void* const* d_in, const int* in_sizes, int n_in,
                              void* d_out, int out_size, void* d_ws, size_t ws_size,
                              hipStream_t stream)
{
  (void)in_sizes; (void)n_in; (void)out_size; (void)ws_size;
  const float* pred  = (const float*)d_in[0];
  const float* trans = (const float*)d_in[1];
  const int*   ref   = (const int*)d_in[2];
  const int*   slen  = (const int*)d_in[3];
  float* ws  = (float*)d_ws;    // [0,256): all-paths scores; [256,512): gold
  float* out = (float*)d_out;

  crf_forward_kernel<<<dim3(16),  dim3(160), 0, stream>>>(pred, trans, slen, ws);
  crf_gold_kernel  <<<dim3(256), dim3(256), 0, stream>>>(pred, trans, ref, slen, ws + 256);
  crf_reduce_kernel<<<dim3(1),   dim3(256), 0, stream>>>(ws, out);
}
